// MixtralSparTAMoeBlock_16990890623335
// MI455X (gfx1250) — matrix operand order repaired and checked by reading, not compiled, not run on hardware
//
#include <hip/hip_runtime.h>

#define T_TOK 1024
#define H_DIM 1024
#define F_DIM 3584
#define E_NUM 8

typedef __attribute__((ext_vector_type(16))) __bf16 v16bf;
typedef __attribute__((ext_vector_type(2)))  __bf16 v2bf;
typedef __attribute__((ext_vector_type(2)))  float  v2f;
typedef __attribute__((ext_vector_type(8)))  float  v8f;

union BFrag  { unsigned u[8]; uint4 q[2]; v16bf v; };

__device__ __forceinline__ unsigned pkbf(float a, float b) {
#if __has_builtin(__builtin_amdgcn_cvt_pk_bf16_f32)
    v2bf t = __builtin_amdgcn_cvt_pk_bf16_f32(a, b);
#else
    v2f f; f[0] = a; f[1] = b;
    v2bf t = __builtin_convertvector(f, v2bf);       // lowers to v_cvt_pk_bf16_f32
#endif
    return __builtin_bit_cast(unsigned, t);
}

__device__ __forceinline__ unsigned short bf16of(float v) {
    return (unsigned short)(pkbf(v, v) & 0xffffu);   // single hw cvt
}

__device__ __forceinline__ uint4 pk8(float4 u, float4 v) {
    uint4 w;
    w.x = pkbf(u.x, u.y); w.y = pkbf(u.z, u.w);
    w.z = pkbf(v.x, v.y); w.w = pkbf(v.z, v.w);
    return w;
}

// fast silu(h1)*h3 -> bf16 (result is bf16-rounded anyway; v_exp/v_rcp suffice)
__device__ __forceinline__ unsigned short silu_mul_bf16(float h1, float h3) {
    float s = __builtin_amdgcn_rcpf(1.f + __expf(-h1));
    return bf16of(h1 * s * h3);
}

// A fragment from a bf16 row in global (K-contig): chunks k0..k0+7, k0+16..k0+23.
__device__ __forceinline__ v16bf load_a_bf16g(const unsigned short* __restrict__ row, int k0) {
    BFrag f;
    f.q[0] = *(const uint4*)(row + k0);
    f.q[1] = *(const uint4*)(row + k0 + 16);
    return f.v;
}

// Fragment from LDS: lane's 32 bytes at base + lane*2 (uint4 units).
__device__ __forceinline__ v16bf frag_lds(const uint4* base, int lane) {
    BFrag f;
    f.q[0] = base[lane * 2];
    f.q[1] = base[lane * 2 + 1];
    return f.v;
}

__device__ __forceinline__ v8f wmma_bf16(v16bf a, v16bf b, v8f c) {
    return __builtin_amdgcn_wmma_f32_16x16x32_bf16(false, a, false, b, (short)0, c, false, false);
}

// ---------------------------------------------------------------- kernels ---

__global__ void zero_kernel(float* __restrict__ out, int n, int* __restrict__ cnt) {
    int i = blockIdx.x * blockDim.x + threadIdx.x;
    for (; i < n; i += gridDim.x * blockDim.x) out[i] = 0.f;
    if (blockIdx.x == 0 && threadIdx.x < E_NUM) cnt[threadIdx.x] = 0;
}

__global__ void router_kernel(const float* __restrict__ x, const float* __restrict__ gw,
                              float* __restrict__ logits_out, int* __restrict__ cnt,
                              int* __restrict__ tok, float* __restrict__ cw) {
    int t = blockIdx.x;
    int lane = threadIdx.x;      // 32 threads = 1 wave
    float acc[E_NUM];
#pragma unroll
    for (int e = 0; e < E_NUM; e++) acc[e] = 0.f;
    const float* xr = x + (size_t)t * H_DIM;
    for (int i = lane; i < H_DIM; i += 32) {
        float xv = xr[i];
#pragma unroll
        for (int e = 0; e < E_NUM; e++) acc[e] += xv * gw[e * H_DIM + i];
    }
#pragma unroll
    for (int e = 0; e < E_NUM; e++)
        for (int o = 16; o > 0; o >>= 1) acc[e] += __shfl_down(acc[e], o, 32);
    if (lane == 0) {
        float mx = acc[0];
#pragma unroll
        for (int e = 1; e < E_NUM; e++) mx = fmaxf(mx, acc[e]);
        float p[E_NUM]; float s = 0.f;
#pragma unroll
        for (int e = 0; e < E_NUM; e++) { p[e] = expf(acc[e] - mx); s += p[e]; }
        int i1 = 0;
#pragma unroll
        for (int e = 1; e < E_NUM; e++) if (p[e] > p[i1]) i1 = e;
        int i2 = (i1 == 0) ? 1 : 0;
#pragma unroll
        for (int e = 0; e < E_NUM; e++) if (e != i1 && p[e] > p[i2]) i2 = e;
        float p1 = p[i1] / s, p2 = p[i2] / s;
        float r1 = p1 / (p1 + p2), r2 = p2 / (p1 + p2);
#pragma unroll
        for (int e = 0; e < E_NUM; e++) logits_out[t * E_NUM + e] = acc[e];
        int s1 = atomicAdd(&cnt[i1], 1); tok[i1 * T_TOK + s1] = t; cw[i1 * T_TOK + s1] = r1;
        int s2 = atomicAdd(&cnt[i2], 1); tok[i2 * T_TOK + s2] = t; cw[i2 * T_TOK + s2] = r2;
    }
}

__global__ void offsets_kernel(const int* __restrict__ cnt, int* __restrict__ offs) {
    if (threadIdx.x == 0) {
        int a = 0;
        for (int e = 0; e < E_NUM; e++) { offs[e] = a; a += cnt[e]; }
        offs[E_NUM] = a;
    }
}

// h1 = X_e W1^T, h3 = X_e W3^T, act = silu(h1)*h3 -> bf16 (compact rows).
// grid: (F/64, T/64, E), block 256 (8 waves). Block tile 64x64, wave tile 16x32.
// Double-buffered LDS staging (fp32->bf16 once per block-step), 1 barrier / 32-K.
__global__ __launch_bounds__(256) void ffn13_kernel(
    const float* __restrict__ x, const float* __restrict__ w1, const float* __restrict__ w3,
    const int* __restrict__ cnt, const int* __restrict__ offs,
    const int* __restrict__ tok, unsigned short* __restrict__ act) {
    const int e = blockIdx.z;
    const int count = cnt[e];
    const int mBase = blockIdx.y * 64;
    if (mBase >= count) return;
    const int nBase = blockIdx.x * 64;
    const int tid  = threadIdx.x;
    const int lane = tid & 31;
    const int wid  = tid >> 5;
    const int wm   = wid >> 1;        // 0..3: 16-row strip
    const int wn   = wid & 1;         // 0..1: 32-col strip
    const int hsel = lane >> 4;

    __shared__ uint4 lds[1536];       // 2 buffers x (A:256 | B1:256 | B3:256) = 24 KB
    uint4* buf0 = lds;
    uint4* buf1 = lds + 768;

    // ---- producer roles: row/col = tid>>2, kgroup = tid&3 (8 K each)
    const int prow = tid >> 2;
    const int pkg  = tid & 3;
    int prA = mBase + prow;
    int prC = (prA < count) ? prA : 0;
    const float* xrow  = x + (size_t)tok[e * T_TOK + prC] * H_DIM + pkg * 8;
    const float* w1row = w1 + ((size_t)e * F_DIM + nBase + prow) * H_DIM + pkg * 8;
    const float* w3row = w3 + ((size_t)e * F_DIM + nBase + prow) * H_DIM + pkg * 8;
    const int aIdx = (prow >> 4) * 64 + (((prow & 15) + 16 * (pkg & 1)) * 2) + (pkg >> 1);
    const int bIdx = (prow >> 4) * 64 + (((prow & 15) + 16 * (pkg & 1)) * 2) + (pkg >> 1);

    const int aOff  = wm * 64;                     // consumer fragment offsets
    const int b1Off = 256 + (wn * 2) * 64;
    const int b3Off = 512 + (wn * 2) * 64;

    v8f a10 = {}, a11 = {}, a30 = {}, a31 = {};

#define FFN13_PRODUCE(dst, koff)                                              \
    {   const float4* pA = (const float4*)(xrow  + (koff));                   \
        const float4* p1 = (const float4*)(w1row + (koff));                   \
        const float4* p3 = (const float4*)(w3row + (koff));                   \
        (dst)[aIdx]       = pk8(pA[0], pA[1]);                                \
        (dst)[256 + bIdx] = pk8(p1[0], p1[1]);                                \
        (dst)[512 + bIdx] = pk8(p3[0], p3[1]); }

#define FFN13_CONSUME(src)                                                    \
    {   v16bf af  = frag_lds((src) + aOff,       lane);                       \
        v16bf b10 = frag_lds((src) + b1Off,      lane);                       \
        v16bf b11 = frag_lds((src) + b1Off + 64, lane);                       \
        v16bf b30 = frag_lds((src) + b3Off,      lane);                       \
        v16bf b31 = frag_lds((src) + b3Off + 64, lane);                       \
        a10 = wmma_bf16(af, b10, a10);                                        \
        a11 = wmma_bf16(af, b11, a11);                                        \
        a30 = wmma_bf16(af, b30, a30);                                        \
        a31 = wmma_bf16(af, b31, a31); }

    FFN13_PRODUCE(buf0, 0)
    for (int kk = 0; kk < H_DIM; kk += 64) {
        // sub-step A: consume buf0 (tile kk), prefetch kk+32 -> buf1
        __syncthreads();
        float4 a0, a1, q10, q11, q30, q31;
        {   const float4* pA = (const float4*)(xrow  + kk + 32); a0  = pA[0]; a1  = pA[1];
            const float4* p1 = (const float4*)(w1row + kk + 32); q10 = p1[0]; q11 = p1[1];
            const float4* p3 = (const float4*)(w3row + kk + 32); q30 = p3[0]; q31 = p3[1]; }
        FFN13_CONSUME(buf0)
        buf1[aIdx]       = pk8(a0,  a1);
        buf1[256 + bIdx] = pk8(q10, q11);
        buf1[512 + bIdx] = pk8(q30, q31);
        // sub-step B: consume buf1 (tile kk+32), prefetch kk+64 -> buf0 (guarded)
        __syncthreads();
        bool more = (kk + 64) < H_DIM;            // uniform
        if (more) {
            const float4* pA = (const float4*)(xrow  + kk + 64); a0  = pA[0]; a1  = pA[1];
            const float4* p1 = (const float4*)(w1row + kk + 64); q10 = p1[0]; q11 = p1[1];
            const float4* p3 = (const float4*)(w3row + kk + 64); q30 = p3[0]; q31 = p3[1];
        }
        FFN13_CONSUME(buf1)
        if (more) {
            buf0[aIdx]       = pk8(a0,  a1);
            buf0[256 + bIdx] = pk8(q10, q11);
            buf0[512 + bIdx] = pk8(q30, q31);
        }
    }

    const int ln = lane & 15;
    const int f0 = nBase + wn * 32 + ln;
    const int f1 = f0 + 16;
    const int rowBase = offs[e];
#pragma unroll
    for (int i = 0; i < 8; i++) {
        int M = i + 8 * hsel;                 // C layout: lane>>4 selects M half
        int r = mBase + wm * 16 + M;
        if (r < count) {
            size_t rowOff = (size_t)(rowBase + r) * F_DIM;
            act[rowOff + f0] = silu_mul_bf16(a10[i], a30[i]);
            act[rowOff + f1] = silu_mul_bf16(a11[i], a31[i]);
        }
    }
}

// out[t,:] += cw * (act_row · W2^T). grid: (H/64, T/64, E), block 256.
// act (bf16) loads straight from global; W2 double-buffered through LDS as bf16.
__global__ __launch_bounds__(256) void ffn2_kernel(
    const unsigned short* __restrict__ act, const float* __restrict__ w2,
    const int* __restrict__ cnt, const int* __restrict__ offs,
    const int* __restrict__ tok, const float* __restrict__ cw,
    float* __restrict__ out) {
    const int e = blockIdx.z;
    const int count = cnt[e];
    const int mBase = blockIdx.y * 64;
    if (mBase >= count) return;
    const int hBase = blockIdx.x * 64;
    const int tid  = threadIdx.x;
    const int lane = tid & 31;
    const int wid  = tid >> 5;
    const int wm   = wid >> 1;
    const int wn   = wid & 1;
    const int hsel = lane >> 4;

    __shared__ uint4 ldsB[512];       // 2 x (64x32 bf16 tile) = 8 KB
    uint4* buf0 = ldsB;
    uint4* buf1 = ldsB + 256;

    const int pcol = tid >> 2;
    const int pkg  = tid & 3;
    const float* w2row = w2 + ((size_t)e * H_DIM + hBase + pcol) * F_DIM + pkg * 8;
    const int bIdx = (pcol >> 4) * 64 + (((pcol & 15) + 16 * (pkg & 1)) * 2) + (pkg >> 1);

    int rA = mBase + wm * 16 + (lane & 15);
    int rC = (rA < count) ? rA : 0;
    const unsigned short* arow = act + (size_t)(offs[e] + rC) * F_DIM;
    const int bOff = (wn * 2) * 64;
    const int ka = hsel * 8;

    v8f c0 = {}, c1 = {};

#define FFN2_CONSUME(src, koff)                                               \
    {   v16bf af = load_a_bf16g(arow, (koff) + ka);                           \
        v16bf b0 = frag_lds((src) + bOff,      lane);                         \
        v16bf b1 = frag_lds((src) + bOff + 64, lane);                         \
        c0 = wmma_bf16(af, b0, c0);                                           \
        c1 = wmma_bf16(af, b1, c1); }

    {   const float4* p = (const float4*)(w2row);
        buf0[bIdx] = pk8(p[0], p[1]); }
    for (int kk = 0; kk < F_DIM; kk += 64) {
        __syncthreads();
        float4 q0, q1;
        {   const float4* p = (const float4*)(w2row + kk + 32); q0 = p[0]; q1 = p[1]; }
        FFN2_CONSUME(buf0, kk)
        buf1[bIdx] = pk8(q0, q1);
        __syncthreads();
        bool more = (kk + 64) < F_DIM;            // uniform
        if (more) { const float4* p = (const float4*)(w2row + kk + 64); q0 = p[0]; q1 = p[1]; }
        FFN2_CONSUME(buf1, kk + 32)
        if (more) buf0[bIdx] = pk8(q0, q1);
    }

    const int ln = lane & 15;
    const int h0 = hBase + wn * 32 + ln;
    const int h1c = h0 + 16;
#pragma unroll
    for (int i = 0; i < 8; i++) {
        int M = i + 8 * hsel;
        int r = mBase + wm * 16 + M;
        if (r < count) {
            int token = tok[e * T_TOK + r];
            float w = cw[e * T_TOK + r];
            unsafeAtomicAdd(&out[(size_t)token * H_DIM + h0],  w * c0[i]);
            unsafeAtomicAdd(&out[(size_t)token * H_DIM + h1c], w * c1[i]);
        }
    }
}

// ----------------------------------------------------------------- launch ---

extern "C" void kernel_launch(void* const* d_in, const int* in_sizes, int n_in,
                              void* d_out, int out_size, void* d_ws, size_t ws_size,
                              hipStream_t stream) {
    (void)in_sizes; (void)n_in; (void)out_size; (void)ws_size;
    const float* x  = (const float*)d_in[0];   // [T, H]
    const float* gw = (const float*)d_in[1];   // [E, H]
    const float* w1 = (const float*)d_in[2];   // [E, F, H]
    const float* w2 = (const float*)d_in[3];   // [E, H, F]
    const float* w3 = (const float*)d_in[4];   // [E, F, H]
    float* out    = (float*)d_out;                       // [T, H]
    float* logits = out + (size_t)T_TOK * H_DIM;         // [T, E]

    char* ws = (char*)d_ws;
    int*   cnt  = (int*)ws;                              // E ints
    int*   offs = (int*)(ws + 32);                       // E+1 ints
    int*   tok  = (int*)(ws + 128);                      // E*T ints
    float* cwv  = (float*)(ws + 128 + 4 * E_NUM * T_TOK);
    unsigned short* act = (unsigned short*)(ws + 128 + 8 * E_NUM * T_TOK);  // 2T x F bf16

    zero_kernel<<<1024, 256, 0, stream>>>(out, T_TOK * H_DIM, cnt);
    router_kernel<<<T_TOK, 32, 0, stream>>>(x, gw, logits, cnt, tok, cwv);
    offsets_kernel<<<1, 32, 0, stream>>>(cnt, offs);
    dim3 g13(F_DIM / 64, T_TOK / 64, E_NUM);
    ffn13_kernel<<<g13, 256, 0, stream>>>(x, w1, w3, cnt, offs, tok, act);
    dim3 g2(H_DIM / 64, T_TOK / 64, E_NUM);
    ffn2_kernel<<<g2, 256, 0, stream>>>(act, w2, cnt, offs, tok, cwv, out);
}
